// SwinTransformer_11244224381558
// MI455X (gfx1250) — matrix operand order repaired and checked by reading, not compiled, not run on hardware
//
#include <hip/hip_runtime.h>
#include <hip/hip_bf16.h>

typedef __attribute__((ext_vector_type(16))) __bf16 v16bf;
typedef __attribute__((ext_vector_type(8)))  float  v8f;
typedef __attribute__((ext_vector_type(4)))  int    v4i;

#define T_TOKENS 131072   // B * H * W = 32 * 4096

// ---------- bf16 helpers (manual RNE, buffers stored as ushort) ----------
__device__ __forceinline__ unsigned short f2bf(float x) {
  union { float f; unsigned u; } c; c.f = x;
  unsigned r = c.u + 0x7FFFu + ((c.u >> 16) & 1u);
  return (unsigned short)(r >> 16);
}
__device__ __forceinline__ float bflo(unsigned u) {  // low bf16 of dword -> f32
  union { unsigned u; float f; } c; c.u = u << 16; return c.f;
}
__device__ __forceinline__ float bfhi(unsigned u) {  // high bf16 of dword -> f32
  union { unsigned u; float f; } c; c.u = u & 0xFFFF0000u; return c.f;
}

// ---------- WMMA fragment helpers (contiguous k = 16*(lane/16) + s for A and B, wave32) ----------
union FragU { uint4 q[2]; v16bf v; };

// B: 32x16 bf16 from Wt stored as [N][K] row-major. lane<16: col=lane, K=k0..k0+15;
//    lane>=16: col=lane-16, K=k0+16..k0+31  (contiguous 32B per lane)
__device__ __forceinline__ v16bf load_fragB(const unsigned short* __restrict__ colPtr, int k0) {
  const int half = (threadIdx.x & 31) >> 4;
  FragU f;
  const uint4* p = (const uint4*)(colPtr + k0 + half * 16);
  f.q[0] = p[0];
  f.q[1] = p[1];
  return f.v;
}

// ---------- GEMM core: block = 16 rows x (8 waves * NT * 16) cols ----------
// A panel (16 x K bf16) staged once per block into LDS via the CDNA5 async
// engine (ASYNCcnt), then consumed by all 8 waves as ds_load_b128 fragments.
template<int NT, int K>
__device__ __forceinline__ void gemm_core(const unsigned short* __restrict__ A,
                                          const unsigned short* __restrict__ Wt,
                                          v8f acc[NT]) {
  constexpr int LDK = K + 8;           // +16B row pad: LDS rows start on distinct banks
  __shared__ unsigned short shA[16 * LDK];
  const int lane = threadIdx.x & 31;
  const int wave = threadIdx.x >> 5;
  const int colBase = wave * NT * 16;

  // ---- cooperative async stage of the 16xK A panel ----
  {
    const unsigned short* gbase = A + (size_t)blockIdx.x * 16 * K;
    constexpr int CPR = K / 8;         // 16B chunks per row
    constexpr int CHUNKS = 16 * CPR;
    for (int c = threadIdx.x; c < CHUNKS; c += 256) {
      int row = c / CPR;
      int cc  = c % CPR;
      const unsigned short* gsrc = gbase + (size_t)row * K + cc * 8;
      unsigned short* ldst = &shA[row * LDK + cc * 8];
#if __has_builtin(__builtin_amdgcn_global_load_async_to_lds_b128)
      __builtin_amdgcn_global_load_async_to_lds_b128(
          (__attribute__((address_space(1))) v4i*)gsrc,
          (__attribute__((address_space(3))) v4i*)ldst, 0, 0);
#else
      *(uint4*)ldst = *(const uint4*)gsrc;
#endif
    }
#if __has_builtin(__builtin_amdgcn_global_load_async_to_lds_b128)
#if __has_builtin(__builtin_amdgcn_s_wait_asynccnt)
    __builtin_amdgcn_s_wait_asynccnt(0);
#else
    asm volatile("s_wait_asynccnt 0" ::: "memory");
#endif
#endif
    __syncthreads();
  }

  const unsigned short* arow = &shA[(lane & 15) * LDK];
  const int ahalf = (lane >> 4) << 4;  // 0 or 16
  const unsigned short* bcol[NT];
#pragma unroll
  for (int t = 0; t < NT; ++t)
    bcol[t] = Wt + (size_t)(colBase + t * 16 + (lane & 15)) * K;
#pragma unroll
  for (int t = 0; t < NT; ++t)
#pragma unroll
    for (int i = 0; i < 8; ++i) acc[t][i] = 0.0f;

  for (int k0 = 0; k0 < K; k0 += 32) {
    // A fragment from LDS (two ds_load_b128, conflict-free via row pad)
    FragU fa;
    fa.q[0] = *(const uint4*)(arow + k0 + ahalf);
    fa.q[1] = *(const uint4*)(arow + k0 + ahalf + 8);
    // batch all B-fragment loads so wmma(t) overlaps loads of t+1..NT-1
    v16bf b[NT];
#pragma unroll
    for (int t = 0; t < NT; ++t) b[t] = load_fragB(bcol[t], k0);
#pragma unroll
    for (int t = 0; t < NT; ++t)
      acc[t] = __builtin_amdgcn_wmma_f32_16x16x32_bf16(
          false, fa.v, false, b[t], (short)0, acc[t], false, false);
  }
}

// ---------- weight transpose + bf16 convert:  Wt[n*K+k] = bf16(W[k*N+n]) ----------
__global__ void __launch_bounds__(256) wprep_k(const float* __restrict__ src,
                                               unsigned short* __restrict__ dst,
                                               int K, int N) {
  int i = blockIdx.x * 256 + threadIdx.x;
  if (i >= K * N) return;
  int k = i / N, n = i % N;
  dst[(size_t)n * K + k] = f2bf(src[i]);
}

// ---------- LayerNorm (+ optional width-roll / channel-roll / window partition) ----------
// one wave per token; shifted=1: LN1 path (reads x at rolled source, shifts channels)
__global__ void __launch_bounds__(256) ln_k(const float* __restrict__ x,
                                            const float* __restrict__ g,
                                            const float* __restrict__ bta,
                                            unsigned short* __restrict__ out,
                                            int shifted) {
  __shared__ unsigned short sh[8 * 256];
  const int lane = threadIdx.x & 31;
  const int wave = threadIdx.x >> 5;
  const int token = blockIdx.x * 8 + wave;
  int srcRow;
  if (shifted) {
    int b = token >> 12, rem = token & 4095, wi = rem >> 2, n = rem & 3;
    int hb = wi >> 5, wb = wi & 31;
    int h = hb * 2 + (n >> 1), w = wb * 2 + (n & 1);
    srcRow = b * 4096 + h * 64 + ((w + 1) & 63);   // jnp.roll(h, -1, axis=2)
  } else {
    srcRow = token;
  }
  const float* src = x + (size_t)srcRow * 256 + lane * 8;
  float4 a0 = ((const float4*)src)[0];
  float4 a1 = ((const float4*)src)[1];
  float va[8] = {a0.x, a0.y, a0.z, a0.w, a1.x, a1.y, a1.z, a1.w};
  float s = 0.f, ss = 0.f;
#pragma unroll
  for (int e = 0; e < 8; ++e) { s += va[e]; ss += va[e] * va[e]; }
#pragma unroll
  for (int m = 16; m >= 1; m >>= 1) {
    s  += __shfl_xor(s,  m, 32);
    ss += __shfl_xor(ss, m, 32);
  }
  float mu = s * (1.f / 256.f);
  float var = ss * (1.f / 256.f) - mu * mu;
  float rs = rsqrtf(var + 1e-5f);
#pragma unroll
  for (int e = 0; e < 8; ++e) {
    int sc = lane * 8 + e;
    sh[wave * 256 + sc] = f2bf((va[e] - mu) * rs * g[sc] + bta[sc]);
  }
  __syncthreads();
  const int off = shifted ? 1 : 0;                 // jnp.roll(h, -1, axis=3)
  unsigned short o[8];
#pragma unroll
  for (int e = 0; e < 8; ++e) {
    int oc = lane * 8 + e;
    o[e] = sh[wave * 256 + ((oc + off) & 255)];
  }
  uint4 pk;
  pk.x = (unsigned)o[0] | ((unsigned)o[1] << 16);
  pk.y = (unsigned)o[2] | ((unsigned)o[3] << 16);
  pk.z = (unsigned)o[4] | ((unsigned)o[5] << 16);
  pk.w = (unsigned)o[6] | ((unsigned)o[7] << 16);
  *(uint4*)(out + (size_t)token * 256 + lane * 8) = pk;
}

// ---------- QKV GEMM: [T,256]x[256,768], bias + q-scale, bf16 out ----------
__global__ void __launch_bounds__(256) gemm_qkv_k(const unsigned short* __restrict__ A,
                                                  const unsigned short* __restrict__ Wt,
                                                  const float* __restrict__ bias,
                                                  unsigned short* __restrict__ out) {
  v8f acc[6];
  gemm_core<6, 256>(A, Wt, acc);
  const int lane = threadIdx.x & 31;
  const int colBase = (threadIdx.x >> 5) * 96;
  const int row0 = blockIdx.x * 16 + ((lane >> 4) << 3);
  const int c0 = lane & 15;
#pragma unroll
  for (int t = 0; t < 6; ++t) {
    int col = colBase + t * 16 + c0;
    float b = bias[col];
    float scale = (col < 256) ? 0.17677669529663687f : 1.0f;  // HD^-0.5
#pragma unroll
    for (int v = 0; v < 8; ++v)
      out[(size_t)(row0 + v) * 768 + col] = f2bf((acc[t][v] + b) * scale);
  }
}

// ---------- windowed attention (N=4, NH=8, HD=32): thread per (win,head,token) ----------
__global__ void __launch_bounds__(256) attn_k(const unsigned short* __restrict__ qkv,
                                              const float* __restrict__ rpb,
                                              unsigned short* __restrict__ out) {
  int tid = blockIdx.x * 256 + threadIdx.x;
  int win = tid >> 5, head = (tid >> 2) & 7, n = tid & 3;
  const size_t base = (size_t)win * 4 * 768 + head * 32;
  float q[32];
  {
    const unsigned* p = (const unsigned*)(qkv + base + (size_t)n * 768);
#pragma unroll
    for (int i = 0; i < 16; ++i) { unsigned u = p[i]; q[2 * i] = bflo(u); q[2 * i + 1] = bfhi(u); }
  }
  float sc[4];
#pragma unroll
  for (int m = 0; m < 4; ++m) {
    const unsigned* p = (const unsigned*)(qkv + base + (size_t)m * 768 + 256);
    float s = 0.f;
#pragma unroll
    for (int i = 0; i < 16; ++i) { unsigned u = p[i]; s += q[2 * i] * bflo(u) + q[2 * i + 1] * bfhi(u); }
    int idx = 3 * ((n >> 1) - (m >> 1) + 1) + ((n & 1) - (m & 1) + 1);  // REL_IDX for WS=2
    sc[m] = s + rpb[idx * 8 + head];
  }
  float mx = fmaxf(fmaxf(sc[0], sc[1]), fmaxf(sc[2], sc[3]));
  float e[4], sum = 0.f;
#pragma unroll
  for (int m = 0; m < 4; ++m) { e[m] = __expf(sc[m] - mx); sum += e[m]; }
  float inv = 1.0f / sum;
  float o[32];
#pragma unroll
  for (int d = 0; d < 32; ++d) o[d] = 0.f;
#pragma unroll
  for (int m = 0; m < 4; ++m) {
    const unsigned* p = (const unsigned*)(qkv + base + (size_t)m * 768 + 512);
    float w = e[m] * inv;
#pragma unroll
    for (int i = 0; i < 16; ++i) { unsigned u = p[i]; o[2 * i] += w * bflo(u); o[2 * i + 1] += w * bfhi(u); }
  }
  unsigned* op = (unsigned*)(out + (size_t)(win * 4 + n) * 256 + head * 32);
#pragma unroll
  for (int i = 0; i < 16; ++i)
    op[i] = (unsigned)f2bf(o[2 * i]) | ((unsigned)f2bf(o[2 * i + 1]) << 16);
}

// ---------- proj GEMM fused with window-reverse + reverse-roll + residual ----------
__global__ void __launch_bounds__(256) gemm_proj_k(const unsigned short* __restrict__ A,
                                                   const unsigned short* __restrict__ Wt,
                                                   const float* __restrict__ bias,
                                                   const float* __restrict__ xin,
                                                   float* __restrict__ x1) {
  v8f acc[2];
  gemm_core<2, 256>(A, Wt, acc);
  const int lane = threadIdx.x & 31;
  const int colBase = (threadIdx.x >> 5) * 32;
  const int row0 = blockIdx.x * 16 + ((lane >> 4) << 3);
  const int c0 = lane & 15;
  int dstrow[8];
#pragma unroll
  for (int v = 0; v < 8; ++v) {
    int tk = row0 + v;
    int b = tk >> 12, rem = tk & 4095, wi = rem >> 2, n = rem & 3;
    int hb = wi >> 5, wb = wi & 31;
    int h = hb * 2 + (n >> 1), w = wb * 2 + (n & 1);
    dstrow[v] = b * 4096 + ((h + 1) & 63) * 64 + ((w + 1) & 63);  // jnp.roll(+1,+1)
  }
#pragma unroll
  for (int t = 0; t < 2; ++t) {
    int col = colBase + t * 16 + c0;
    float b = bias[col];
#pragma unroll
    for (int v = 0; v < 8; ++v) {
      size_t di = (size_t)dstrow[v] * 256 + col;
      x1[di] = xin[di] + acc[t][v] + b;
    }
  }
}

// ---------- MLP GEMM1: [T,256]x[256,1024] + bias + exact GELU -> bf16 ----------
__global__ void __launch_bounds__(256) gemm_mlp1_k(const unsigned short* __restrict__ A,
                                                   const unsigned short* __restrict__ Wt,
                                                   const float* __restrict__ bias,
                                                   unsigned short* __restrict__ out) {
  v8f acc[8];
  gemm_core<8, 256>(A, Wt, acc);
  const int lane = threadIdx.x & 31;
  const int colBase = (threadIdx.x >> 5) * 128;
  const int row0 = blockIdx.x * 16 + ((lane >> 4) << 3);
  const int c0 = lane & 15;
#pragma unroll
  for (int t = 0; t < 8; ++t) {
    int col = colBase + t * 16 + c0;
    float b = bias[col];
#pragma unroll
    for (int v = 0; v < 8; ++v) {
      float u = acc[t][v] + b;
      float ge = 0.5f * u * (1.0f + erff(u * 0.70710678118654752f));
      out[(size_t)(row0 + v) * 1024 + col] = f2bf(ge);
    }
  }
}

// ---------- MLP GEMM2: [T,1024]x[1024,256] + bias + residual -> d_out (fp32) ----------
__global__ void __launch_bounds__(256) gemm_mlp2_k(const unsigned short* __restrict__ A,
                                                   const unsigned short* __restrict__ Wt,
                                                   const float* __restrict__ bias,
                                                   const float* __restrict__ x1,
                                                   float* __restrict__ out) {
  v8f acc[2];
  gemm_core<2, 1024>(A, Wt, acc);
  const int lane = threadIdx.x & 31;
  const int colBase = (threadIdx.x >> 5) * 32;
  const int row0 = blockIdx.x * 16 + ((lane >> 4) << 3);
  const int c0 = lane & 15;
#pragma unroll
  for (int t = 0; t < 2; ++t) {
    int col = colBase + t * 16 + c0;
    float b = bias[col];
#pragma unroll
    for (int v = 0; v < 8; ++v) {
      size_t di = (size_t)(row0 + v) * 256 + col;
      out[di] = x1[di] + acc[t][v] + b;
    }
  }
}

extern "C" void kernel_launch(void* const* d_in, const int* in_sizes, int n_in,
                              void* d_out, int out_size, void* d_ws, size_t ws_size,
                              hipStream_t stream) {
  const float* x      = (const float*)d_in[0];
  const float* n1g    = (const float*)d_in[1];
  const float* n1b    = (const float*)d_in[2];
  const float* qkv_w  = (const float*)d_in[3];
  const float* qkv_b  = (const float*)d_in[4];
  const float* proj_w = (const float*)d_in[5];
  const float* proj_b = (const float*)d_in[6];
  const float* rpb    = (const float*)d_in[7];
  const float* n2g    = (const float*)d_in[8];
  const float* n2b    = (const float*)d_in[9];
  const float* w1     = (const float*)d_in[10];
  const float* b1     = (const float*)d_in[11];
  const float* w2     = (const float*)d_in[12];
  const float* b2     = (const float*)d_in[13];

  char* ws = (char*)d_ws;
  // [0, 64M)      win_a bf16 [T,256]   (reused later as h2)
  // [64M, 256M)   qkv   bf16 [T,768]   (reused later as mlp hidden, 256MB with attn)
  // [256M, 320M)  attn  bf16 [T,256]
  // [320M, 448M)  x1    fp32 [T,256]
  // [448M+, ...)  bf16 transposed weights
  unsigned short* win_a  = (unsigned short*)(ws);
  unsigned short* qkvb   = (unsigned short*)(ws + 67108864ull);
  unsigned short* attn_o = (unsigned short*)(ws + 268435456ull);
  float*          x1     = (float*)        (ws + 335544320ull);
  unsigned short* h2     = win_a;                                   // reuse
  unsigned short* hid    = (unsigned short*)(ws + 67108864ull);     // reuse qkv+attn
  unsigned short* qkv_wt = (unsigned short*)(ws + 469762048ull);
  unsigned short* proj_wt = qkv_wt + 768 * 256;
  unsigned short* w1t     = proj_wt + 256 * 256;
  unsigned short* w2t     = w1t + 1024 * 256;

  dim3 blk(256);
  wprep_k<<<(256 * 768 + 255) / 256, blk, 0, stream>>>(qkv_w, qkv_wt, 256, 768);
  wprep_k<<<(256 * 256 + 255) / 256, blk, 0, stream>>>(proj_w, proj_wt, 256, 256);
  wprep_k<<<(256 * 1024 + 255) / 256, blk, 0, stream>>>(w1, w1t, 256, 1024);
  wprep_k<<<(1024 * 256 + 255) / 256, blk, 0, stream>>>(w2, w2t, 1024, 256);

  ln_k<<<T_TOKENS / 8, blk, 0, stream>>>(x, n1g, n1b, win_a, 1);
  gemm_qkv_k<<<T_TOKENS / 16, blk, 0, stream>>>(win_a, qkv_wt, qkv_b, qkvb);
  attn_k<<<(T_TOKENS * 8) / 256, blk, 0, stream>>>(qkvb, rpb, attn_o);
  gemm_proj_k<<<T_TOKENS / 16, blk, 0, stream>>>(attn_o, proj_wt, proj_b, x, x1);
  ln_k<<<T_TOKENS / 8, blk, 0, stream>>>(x1, n2g, n2b, h2, 0);
  gemm_mlp1_k<<<T_TOKENS / 16, blk, 0, stream>>>(h2, w1t, b1, hid);
  gemm_mlp2_k<<<T_TOKENS / 16, blk, 0, stream>>>(hid, w2t, b2, x1, (float*)d_out);
}
